// Temporal_Brain_Layer_4853313044826
// MI455X (gfx1250) — matrix operand order repaired and checked by reading, not compiled, not run on hardware
//
#include <hip/hip_runtime.h>

// Fused room-masked multi-head attention for MI455X (gfx1250, wave32, WMMA).
// Kernel 1: pack Wqkv/Wout (f32) into f16 WMMA B-fragment order in workspace.
// Kernel 2: one workgroup (256 threads = 8 waves) per attention instance;
//           all GEMMs via v_wmma_f32_16x16x32_f16, B matrices streamed from
//           pre-packed L2-resident fragments, intermediates in LDS (48KB).

typedef __attribute__((ext_vector_type(16))) _Float16 v16h;
typedef __attribute__((ext_vector_type(8))) float v8f;

#define S_TOK 26
#define SP    32
#define E_DIM 128
#define E3    384

// Fragment-array sizes (elements of _Float16)
#define QKV_FRAG_ELEMS (4 * 24 * 32 * 16)   // ksteps * ntiles * lanes * 16
#define OUT_FRAG_ELEMS (4 * 8 * 32 * 16)

__constant__ int d_room[25] = {5,5,5,5,1,1,1,1,2,2,2,2,3,3,3,3,4,4,4,4,5,1,1,2,2};

// A-matrix (16x32 f16): lane half owns a contiguous K range of 16 (k = 16*(lane>=16)+e).
__device__ __forceinline__ int a_k0(int j, int lane) {
  return ((lane >> 4) << 4) + (j << 1);
}
// B-matrix (32x16 f16): lane half owns a contiguous K range of 16 (k = 16*(lane>=16)+e).
__device__ __forceinline__ int b_k0(int j, int lane) {
  return ((lane >> 4) << 4) + (j << 1);
}

__device__ __forceinline__ v16h load_a_frag(const _Float16* lanerow, int kbase, int lane) {
  union { v16h v; unsigned u[8]; } r;
#pragma unroll
  for (int j = 0; j < 8; ++j)
    r.u[j] = *(const unsigned*)(lanerow + kbase + a_k0(j, lane));
  return r.v;
}

__device__ __forceinline__ v16h load_a_frag_pad(const _Float16* lanerow, int kmax, int lane) {
  union { v16h v; unsigned u[8]; } r;
#pragma unroll
  for (int j = 0; j < 8; ++j) {
    int k = a_k0(j, lane);
    r.u[j] = (k < kmax) ? *(const unsigned*)(lanerow + k) : 0u;
  }
  return r.v;
}

__device__ __forceinline__ v16h load_b_frag_lds(const _Float16* lanecol, int kmax, int lane) {
  union { v16h v; unsigned u[8]; } r;
#pragma unroll
  for (int j = 0; j < 8; ++j) {
    int k = b_k0(j, lane);
    r.u[j] = (k < kmax) ? *(const unsigned*)(lanecol + k) : 0u;
  }
  return r.v;
}

__device__ __forceinline__ v16h load_b_frag_strided(const _Float16* base, int strideK,
                                                    int kmax, int lane) {
  union { v16h v; _Float16 h[16]; } r;
#pragma unroll
  for (int j = 0; j < 8; ++j) {
    int k = b_k0(j, lane);
    r.h[2*j]   = (k     < kmax) ? base[k * strideK]       : (_Float16)0.f;
    r.h[2*j+1] = (k + 1 < kmax) ? base[(k + 1) * strideK] : (_Float16)0.f;
  }
  return r.v;
}

__device__ __forceinline__ bool mask_ok(int q, int k) {
  if (q >= S_TOK || k >= S_TOK) return false;      // padded rows/cols
  if (q == 25 || k == 25) return true;             // global token
  return d_room[q] == d_room[k];                   // same room
}

// ---------------------------------------------------------------------------
// Kernel 1: pack weights into B-fragment order (f16).
// wqkvF[((kstep*24 + ntile)*32 + lane)*16 + e] = Wqkv[kstep*32 + kloc][ntile*16 + n]
//   with kloc = 16*(lane>=16) + e  (contiguous-K B layout), n = lane&15.
// ---------------------------------------------------------------------------
__global__ __launch_bounds__(256) void pack_weights(
    const float* __restrict__ Wqkv, const float* __restrict__ Wout,
    _Float16* __restrict__ wqkvF, _Float16* __restrict__ woutF) {
  int idx = blockIdx.x * 256 + threadIdx.x;   // 65536 total
  int e = idx & 15;
  int lane = (idx >> 4) & 31;
  int kloc = ((lane >> 4) << 4) + e;
  int nl = lane & 15;
  if (idx < QKV_FRAG_ELEMS) {
    int tileIdx = idx >> 9;                   // kstep*24 + ntile
    int kstep = tileIdx / 24;
    int ntile = tileIdx - kstep * 24;
    wqkvF[idx] = (_Float16)Wqkv[(kstep * 32 + kloc) * E3 + ntile * 16 + nl];
  } else {
    int j = idx - QKV_FRAG_ELEMS;
    int tileIdx = j >> 9;                     // kstep*8 + ntile
    int kstep = tileIdx >> 3;
    int ntile = tileIdx & 7;
    woutF[j] = (_Float16)Wout[(kstep * 32 + kloc) * E_DIM + ntile * 16 + nl];
  }
}

// ---------------------------------------------------------------------------
// Kernel 2: fused attention, one instance per workgroup.
// ---------------------------------------------------------------------------
__global__ __launch_bounds__(256) void fused_room_attn(
    const float* __restrict__ x,
    const _Float16* __restrict__ wqkvF, const _Float16* __restrict__ woutF,
    const float* __restrict__ bqkv, const float* __restrict__ bout,
    float* __restrict__ out) {
  __shared__ _Float16 As[SP * E_DIM];    // 8KB  (x tile, later reused as attn tile)
  __shared__ _Float16 QKVs[SP * E3];     // 24KB (q|k|v per token, (3,H,16) cols)
  __shared__ _Float16 Ps[8 * 32 * 32];   // 16KB (softmax weights, one 32x32 per head)

  const int tid  = threadIdx.x;
  const int w    = tid >> 5;       // wave id 0..7
  const int lane = tid & 31;
  const int nloc = lane & 15;
  const int n    = blockIdx.x;     // attention instance
  const long long xbase = (long long)n * (S_TOK * E_DIM);

  const v16h* WQ = (const v16h*)wqkvF;   // [kstep][ntile 0..23][lane]
  const v16h* WO = (const v16h*)woutF;   // [kstep][ntile 0..7 ][lane]

  v8f zero8 = {0.f, 0.f, 0.f, 0.f, 0.f, 0.f, 0.f, 0.f};

  // ---- Stage A: x -> As (f16), rows 26..31 zeroed --------------------------
#pragma unroll
  for (int it = 0; it < 16; ++it) {
    int i = tid + it * 256;                 // i = row*128 + col over padded tile
    int row = i >> 7;
    As[i] = (_Float16)((row < S_TOK) ? x[xbase + i] : 0.f);
  }
  __syncthreads();

  // ---- Stage B: QKV = As(32x128) @ Wqkv(128x384) + bqkv --------------------
  const int mtile = w & 1;         // each wave: fixed M tile, 6 N tiles
  const int q4    = w >> 1;        // 0..3 ; ntile = q4 + 4*jj
  const _Float16* arowB = As + (mtile * 16 + nloc) * E_DIM;

  v8f accQ[6];
#pragma unroll
  for (int i = 0; i < 6; ++i) accQ[i] = zero8;

  for (int kstep = 0; kstep < 4; ++kstep) {
    v16h a = load_a_frag(arowB, kstep * 32, lane);
#pragma unroll
    for (int jj = 0; jj < 6; ++jj) {
      int ntile = q4 + 4 * jj;
      v16h b = WQ[(kstep * 24 + ntile) * 32 + lane];
      if (kstep < 3)  // warm WGP cache for next K-slice (WGP-scope prefetch)
        __builtin_prefetch((const void*)&WQ[((kstep + 1) * 24 + ntile) * 32 + lane], 0, 3);
      accQ[jj] = __builtin_amdgcn_wmma_f32_16x16x32_f16(
          false, a, false, b, (short)0, accQ[jj], false, false);
    }
  }
  // store QKV with bias (f16); rows 26..31 hold pure bias (masked away later)
#pragma unroll
  for (int jj = 0; jj < 6; ++jj) {
    int ntile = q4 + 4 * jj;
    int c = ntile * 16 + nloc;
    float bb = bqkv[c];
#pragma unroll
    for (int r = 0; r < 8; ++r) {
      int m = mtile * 16 + r + ((lane >> 4) << 3);
      QKVs[m * E3 + c] = (_Float16)(accQ[jj][r] + bb);
    }
  }
  __syncthreads();

  // ---- Stage C: per-head attention (wave w == head h) ----------------------
  const int h = w;
  v8f sacc[2][2];
  {
    v16h bkt0 = load_b_frag_lds(QKVs + (0 * 16 + nloc) * E3 + 128 + h * 16, 16, lane);
    v16h bkt1 = load_b_frag_lds(QKVs + (1 * 16 + nloc) * E3 + 128 + h * 16, 16, lane);
#pragma unroll
    for (int mt = 0; mt < 2; ++mt) {
      v16h aq = load_a_frag_pad(QKVs + (mt * 16 + nloc) * E3 + h * 16, 16, lane);
      sacc[mt][0] = __builtin_amdgcn_wmma_f32_16x16x32_f16(
          false, aq, false, bkt0, (short)0, zero8, false, false);
      sacc[mt][1] = __builtin_amdgcn_wmma_f32_16x16x32_f16(
          false, aq, false, bkt1, (short)0, zero8, false, false);
    }
  }
  // masked softmax in registers; a row's 16 cols live across one half-wave
  _Float16* Psw = Ps + h * (32 * 32);
  const float scale = 0.25f;  // 1/sqrt(16)
#pragma unroll
  for (int mt = 0; mt < 2; ++mt) {
#pragma unroll
    for (int r = 0; r < 8; ++r) {
      int m = mt * 16 + r + ((lane >> 4) << 3);
      float s0 = sacc[mt][0][r] * scale;
      float s1 = sacc[mt][1][r] * scale;
      if (!mask_ok(m, nloc))      s0 = -1e30f;
      if (!mask_ok(m, nloc + 16)) s1 = -1e30f;
      float mx = fmaxf(s0, s1);
      for (int d = 1; d < 16; d <<= 1) mx = fmaxf(mx, __shfl_xor(mx, d, 16));
      float e0 = __expf(s0 - mx);
      float e1 = __expf(s1 - mx);
      float sm = e0 + e1;
      for (int d = 1; d < 16; d <<= 1) sm += __shfl_xor(sm, d, 16);
      float inv = 1.0f / sm;
      Psw[m * 32 + nloc]      = (_Float16)(e0 * inv);
      Psw[m * 32 + nloc + 16] = (_Float16)(e1 * inv);
    }
  }
  // attn = P(32x32) @ V(32x16); result overwrites As as the attn tile
  {
    v16h bv = load_b_frag_strided(QKVs + 256 + h * 16 + nloc, E3, 32, lane);
#pragma unroll
    for (int mt = 0; mt < 2; ++mt) {
      v16h ap = load_a_frag(Psw + (mt * 16 + nloc) * 32, 0, lane);
      v8f o = __builtin_amdgcn_wmma_f32_16x16x32_f16(
          false, ap, false, bv, (short)0, zero8, false, false);
#pragma unroll
      for (int r = 0; r < 8; ++r) {
        int m = mt * 16 + r + ((lane >> 4) << 3);
        As[m * E_DIM + h * 16 + nloc] = (_Float16)o[r];
      }
    }
  }
  __syncthreads();

  // ---- Stage D: out = attn(32x128) @ Wout(128x128) + bout ------------------
  v8f accO[2];
  accO[0] = zero8;
  accO[1] = zero8;
  const _Float16* arowD = As + (mtile * 16 + nloc) * E_DIM;
  for (int kstep = 0; kstep < 4; ++kstep) {
    v16h a = load_a_frag(arowD, kstep * 32, lane);
#pragma unroll
    for (int jj = 0; jj < 2; ++jj) {
      int ntile = q4 + 4 * jj;
      v16h b = WO[(kstep * 8 + ntile) * 32 + lane];
      if (kstep < 3)
        __builtin_prefetch((const void*)&WO[((kstep + 1) * 8 + ntile) * 32 + lane], 0, 3);
      accO[jj] = __builtin_amdgcn_wmma_f32_16x16x32_f16(
          false, a, false, b, (short)0, accO[jj], false, false);
    }
  }
#pragma unroll
  for (int jj = 0; jj < 2; ++jj) {
    int ntile = q4 + 4 * jj;
    int c = ntile * 16 + nloc;
    float bb = bout[c];
#pragma unroll
    for (int r = 0; r < 8; ++r) {
      int m = mtile * 16 + r + ((lane >> 4) << 3);
      if (m < S_TOK)
        out[xbase + m * E_DIM + c] = accO[jj][r] + bb;
    }
  }
}

extern "C" void kernel_launch(void* const* d_in, const int* in_sizes, int n_in,
                              void* d_out, int out_size, void* d_ws, size_t ws_size,
                              hipStream_t stream) {
  (void)n_in; (void)out_size; (void)ws_size;
  const float* x    = (const float*)d_in[0];
  const float* Wqkv = (const float*)d_in[1];
  const float* bqkv = (const float*)d_in[2];
  const float* Wout = (const float*)d_in[3];
  const float* bout = (const float*)d_in[4];
  float* out = (float*)d_out;

  _Float16* wqkvF = (_Float16*)d_ws;                 // 96KB
  _Float16* woutF = wqkvF + QKV_FRAG_ELEMS;          // 32KB  (ws needs 128KB)

  const int Ninst = in_sizes[0] / (S_TOK * E_DIM);   // B*T = 8192

  // Pack weights into fragment order (re-done every call: capture-safe,
  // deterministic, and negligible next to the 218MB main-kernel traffic).
  pack_weights<<<dim3((QKV_FRAG_ELEMS + OUT_FRAG_ELEMS) / 256), dim3(256), 0, stream>>>(
      Wqkv, Wout, wqkvF, woutF);

  fused_room_attn<<<dim3(Ninst), dim3(256), 0, stream>>>(
      x, wqkvF, woutF, bqkv, bout, out);
}
